// TemporalDeformableAttention_10445360464345
// MI455X (gfx1250) — matrix operand order repaired and checked by reading, not compiled, not run on hardware
//
#include <hip/hip_runtime.h>
#include <hip/hip_bf16.h>
#include <stdint.h>

// ---- compile-time shapes (reference: B=4,T=8,Lv=9216,Lq=9216,C=256,H=8,K=4) ----
static constexpr int BB = 4;
static constexpr int TT = 8;
static constexpr int LV = 9216;
static constexpr int LQ = 9216;
static constexpr int CC = 256;   // channels (= N = K of both GEMMs)
static constexpr int HH = 8;
static constexpr int KP = 4;     // sampling points
static constexpr int HD = 32;    // head dim

typedef __attribute__((ext_vector_type(16))) __bf16    v16bf;
typedef __attribute__((ext_vector_type(8)))  float     v8f;
typedef __attribute__((ext_vector_type(4)))  float     f32x4;
typedef __attribute__((ext_vector_type(4)))  uint32_t  u32x4;
typedef __attribute__((ext_vector_type(8)))  uint32_t  u32x8;

__device__ __forceinline__ __bf16 f2bf(float f) {
  uint32_t u = __builtin_bit_cast(uint32_t, f);
  u += 0x7FFFu + ((u >> 16) & 1u);               // round-to-nearest-even
  uint16_t h = (uint16_t)(u >> 16);
  return __builtin_bit_cast(__bf16, h);
}
__device__ __forceinline__ float bf2f(uint16_t h) {
  uint32_t u = ((uint32_t)h) << 16;
  return __builtin_bit_cast(float, u);
}

// ---- prep: W[k,n] (f32, [256,256]) -> Wt[n,k] (bf16) ----
__global__ __launch_bounds__(256) void conv_transpose_w(const float* __restrict__ W,
                                                        uint16_t* __restrict__ Wt) {
  int i = blockIdx.x * 256 + threadIdx.x;        // i = k*256 + n  (coalesced read)
  int k = i >> 8, n = i & 255;
  Wt[n * 256 + k] = __builtin_bit_cast(uint16_t, f2bf(W[i]));
}

// ---- bf16-WMMA GEMM: Y[M,256] = f32(X[M,256]) @ bf16(Wt[256,256]^T) + bias ----
// Weights staged to LDS once per block via TDM (tensor_load_to_lds); inner loop is a
// 128-step software-pipelined rotation: B fragment i+1 ds_load'ed before WMMA i, and
// the next kt's A tile global-loaded before the current kt's 16 WMMAs.
template <bool OUT_BF16>
__global__ __launch_bounds__(256) void gemm256_bf16wmma(const float* __restrict__ X,
                                                        const uint16_t* __restrict__ Wt,
                                                        const float* __restrict__ bias,
                                                        void* __restrict__ Y) {
  extern __shared__ uint16_t lds_w[];            // [256][256] bf16 = 128 KB

  // ---- wave 0: TDM load of the whole weight tile into LDS ----
  if (threadIdx.x < 32) {
    const uint64_t ga = (uint64_t)(uintptr_t)Wt;
    const uint32_t lb = (uint32_t)(uintptr_t)lds_w;   // flat addr low 32b == LDS offset
    u32x4 g0;
    g0[0] = 1u;                                  // count=1, no gather, user descriptor
    g0[1] = lb;                                  // lds_addr (bytes)
    g0[2] = (uint32_t)ga;                        // global_addr[31:0]
    g0[3] = (uint32_t)((ga >> 32) & 0x1FFFFFFu)  // global_addr[56:32]
          | (2u << 30);                          // type = 2 ("image")
    u32x8 g1;
    g1[0] = 1u << 16;                            // data_size=1 (2B); wg_mask=0; no pad/iter
    g1[1] = 256u << 16;                          // tensor_dim0[15:0]=256 (bits 63:48)
    g1[2] = 256u << 16;                          // tensor_dim1[15:0]=256 (bits 95:80)
    g1[3] = 256u << 16;                          // tile_dim0=256      (bits 127:112)
    g1[4] = 256u;                                // tile_dim1=256      (bits 143:128)
    g1[5] = 256u;                                // tensor_dim0_stride[31:0]=256
    g1[6] = 0u;
    g1[7] = 0u;
    const u32x4 gz = {0u, 0u, 0u, 0u};           // groups 2/3: unused dims (2-D tile)
    asm volatile("tensor_load_to_lds %0, %1, %2, %3"
                 :: "s"(g0), "s"(g1), "s"(gz), "s"(gz)
                 : "memory");
#if __has_builtin(__builtin_amdgcn_s_wait_tensorcnt)
    __builtin_amdgcn_s_wait_tensorcnt(0);
#else
    asm volatile("s_wait_tensorcnt 0x0" ::: "memory");
#endif
  }
  __syncthreads();                               // publish LDS weight tile to all waves

  const int lane  = threadIdx.x & 31;
  const int wv    = threadIdx.x >> 5;            // 8 waves / block
  const int col   = lane & 15;
  const int kh    = lane >> 4;                   // K-half select: A/B hold k = 16*kh + s (contiguous)
  const long mtile = (long)blockIdx.x * 8 + wv;
  const long row   = mtile * 16 + col;

  v8f acc[16];
  #pragma unroll
  for (int nt = 0; nt < 16; ++nt) {
    const float bn = bias[nt * 16 + col];        // D: N = nt*16+col for all 8 lanes' M rows
    #pragma unroll
    for (int i = 0; i < 8; ++i) acc[nt][i] = bn;
  }

  // B fragment: lane holds col N=col, 16 contiguous K at kh*16 (32B aligned, LDS)
  const uint16_t* bbase = lds_w + col * 256 + kh * 16;

  const float* xr = X + row * 256;
  // prologue: A tile for kt=0, first B fragment
  f32x4 x0 = *(const f32x4*)(xr + kh * 16);
  f32x4 x1 = *(const f32x4*)(xr + kh * 16 + 4);
  f32x4 x2 = *(const f32x4*)(xr + kh * 16 + 8);
  f32x4 x3 = *(const f32x4*)(xr + kh * 16 + 12);
  v16bf bcur = *(const v16bf*)(bbase);

  #pragma unroll
  for (int kt = 0; kt < 8; ++kt) {
    f32x4 y0, y1, y2, y3;
    if (kt < 7) {                                // double-buffer next kt's A tile
      const int nb = (kt + 1) * 32 + kh * 16;
      __builtin_prefetch(xr + (kt + 1) * 32 + 64, 0, 0);
      y0 = *(const f32x4*)(xr + nb);
      y1 = *(const f32x4*)(xr + nb + 4);
      y2 = *(const f32x4*)(xr + nb + 8);
      y3 = *(const f32x4*)(xr + nb + 12);
    }
    v16bf a;
    #pragma unroll
    for (int i = 0; i < 4; ++i) {
      a[i]      = f2bf(x0[i]);
      a[4 + i]  = f2bf(x1[i]);
      a[8 + i]  = f2bf(x2[i]);
      a[12 + i] = f2bf(x3[i]);
    }
    #pragma unroll
    for (int nt = 0; nt < 16; ++nt) {            // rotate B fragments (incl. across kt)
      v16bf bnext = bcur;
      if (nt < 15)      bnext = *(const v16bf*)(bbase + (nt + 1) * 16 * 256 + kt * 32);
      else if (kt < 7)  bnext = *(const v16bf*)(bbase + (kt + 1) * 32);
      acc[nt] = __builtin_amdgcn_wmma_f32_16x16x32_bf16(
          /*neg_a=*/false, a, /*neg_b=*/false, bcur,
          /*c_mod=*/(short)0, acc[nt], /*reuse_a=*/false, /*reuse_b=*/false);
      bcur = bnext;
    }
    x0 = y0; x1 = y1; x2 = y2; x3 = y3;
  }

  #pragma unroll
  for (int nt = 0; nt < 16; ++nt) {
    #pragma unroll
    for (int i = 0; i < 8; ++i) {
      const long R = mtile * 16 + kh * 8 + i;    // D layout: vgpr i -> M = i (+8 for hi half)
      const long Nc = nt * 16 + col;
      if (OUT_BF16)
        ((uint16_t*)Y)[R * 256 + Nc] = __builtin_bit_cast(uint16_t, f2bf(acc[nt][i]));
      else
        ((float*)Y)[R * 256 + Nc] = acc[nt][i];
    }
  }
}

// ---- fused: off/attn projections + softmax + temporal bilinear gather ----
__global__ __launch_bounds__(256) void sampler(const float* __restrict__ q,
                                               const int* __restrict__ idx_q,
                                               const float* __restrict__ tref,
                                               const float* __restrict__ Woff,
                                               const float* __restrict__ boff,
                                               const float* __restrict__ Wattn,
                                               const float* __restrict__ battn,
                                               const uint16_t* __restrict__ Vbf,
                                               float* __restrict__ hout) {
  __shared__ float qs[CC];
  __shared__ float offs[HH * KP], wgt[HH * KP], alph[HH * KP];
  __shared__ float logit[HH * KP];
  __shared__ int   t0s[HH * KP];

  const int bid = blockIdx.x;                    // b*LQ + lq
  const int b   = bid / LQ;
  const int tid = threadIdx.x;

  qs[tid] = q[(size_t)bid * CC + tid];
  __syncthreads();

  if (tid < 64) {                                // 64 dot products of length 256
    const int j = tid & 31;
    const float* W = (tid < 32) ? Woff : Wattn;
    float s = (tid < 32) ? boff[j] : battn[j];
    #pragma unroll 8
    for (int c = 0; c < CC; ++c) s += qs[c] * W[c * 32 + j];
    if (tid < 32) offs[j] = s; else logit[j] = s;
  }
  __syncthreads();

  if (tid < HH) {                                // per-head softmax + sample times
    const int h = tid;
    float m = logit[h * KP];
    #pragma unroll
    for (int k = 1; k < KP; ++k) m = fmaxf(m, logit[h * KP + k]);
    float e[KP], sum = 0.f;
    #pragma unroll
    for (int k = 0; k < KP; ++k) { e[k] = __expf(logit[h * KP + k] - m); sum += e[k]; }
    const float inv = 1.f / sum;
    const float tr = tref[bid];
    #pragma unroll
    for (int k = 0; k < KP; ++k) {
      wgt[h * KP + k] = e[k] * inv;
      float ts = fminf(fmaxf(tr + offs[h * KP + k], 0.f), (float)(TT - 1));
      int t0 = (int)floorf(ts);
      if (t0 > TT - 1) t0 = TT - 1;
      t0s[h * KP + k]  = t0;
      alph[h * KP + k] = ts - (float)t0;
    }
  }
  __syncthreads();

  const int h = tid >> 5, d = tid & 31;
  const int iv = idx_q[bid];
  const size_t base = (size_t)b * TT * LV * CC + (size_t)iv * CC + h * HD + d;
  const size_t tstride = (size_t)LV * CC;

  float acc = 0.f;
  #pragma unroll
  for (int k = 0; k < KP; ++k) {
    const int   t0 = t0s[h * KP + k];
    const int   t1 = (t0 + 1 < TT) ? t0 + 1 : TT - 1;
    const float a  = alph[h * KP + k];
    const float w  = wgt[h * KP + k];
    const float v0 = bf2f(Vbf[base + (size_t)t0 * tstride]);  // 64B coalesced per head group
    const float v1 = bf2f(Vbf[base + (size_t)t1 * tstride]);
    acc += w * ((1.f - a) * v0 + a * v1);
  }
  hout[(size_t)bid * CC + tid] = acc;
}

extern "C" void kernel_launch(void* const* d_in, const int* in_sizes, int n_in,
                              void* d_out, int out_size, void* d_ws, size_t ws_size,
                              hipStream_t stream) {
  (void)in_sizes; (void)n_in; (void)out_size; (void)ws_size;
  const float* Fp      = (const float*)d_in[0];
  const float* queries = (const float*)d_in[1];
  const int*   idx_q   = (const int*)d_in[2];
  const float* tref    = (const float*)d_in[3];
  const float* W_off   = (const float*)d_in[4];
  const float* b_off   = (const float*)d_in[5];
  const float* W_attn  = (const float*)d_in[6];
  const float* b_attn  = (const float*)d_in[7];
  const float* W_v     = (const float*)d_in[8];
  const float* b_v     = (const float*)d_in[9];
  const float* W_out   = (const float*)d_in[10];
  const float* b_out   = (const float*)d_in[11];
  float* out = (float*)d_out;

  // workspace carve-up
  char* ws = (char*)d_ws;
  uint16_t* Vbf = (uint16_t*)ws;                                   // [B,T,Lv,C] bf16
  size_t off = ((size_t)BB * TT * LV * CC * 2 + 255) & ~(size_t)255;
  uint16_t* WvT = (uint16_t*)(ws + off); off += (size_t)CC * CC * 2;
  uint16_t* WoT = (uint16_t*)(ws + off); off += (size_t)CC * CC * 2;
  off = (off + 255) & ~(size_t)255;
  float* hout = (float*)(ws + off);                                // [B,Lq,C] f32

  const size_t lds_bytes = (size_t)CC * CC * sizeof(uint16_t);     // 128 KB weight tile

  conv_transpose_w<<<CC, 256, 0, stream>>>(W_v, WvT);
  conv_transpose_w<<<CC, 256, 0, stream>>>(W_out, WoT);

  // V = Fp @ W_v + b_v  (M = B*T*Lv = 294912), output bf16
  gemm256_bf16wmma<true><<<(BB * TT * LV) / 128, 256, lds_bytes, stream>>>(Fp, WvT, b_v, (void*)Vbf);

  // fused projections + softmax + gather -> head_out f32
  sampler<<<BB * LQ, 256, 0, stream>>>(queries, idx_q, tref,
                                       W_off, b_off, W_attn, b_attn, Vbf, hout);

  // out = head_out @ W_out + b_out  (M = B*Lq = 36864), output f32
  gemm256_bf16wmma<false><<<(BB * LQ) / 128, 256, lds_bytes, stream>>>(hout, WoT, b_out, (void*)out);
}
